// TPAttention_63840393888086
// MI455X (gfx1250) — matrix operand order repaired and checked by reading, not compiled, not run on hardware
//
#include <hip/hip_runtime.h>
#include <hip/hip_bf16.h>

typedef _Float16 h8   __attribute__((ext_vector_type(8)));
typedef _Float16 v16h __attribute__((ext_vector_type(16)));
typedef float    v8f  __attribute__((ext_vector_type(8)));

// ---------------- problem constants ----------------
constexpr int Bk   = 4;
constexpr int Tk   = 1024;
constexpr int Hk   = 4096;
constexpr int NQ   = 32;
constexpr int NKV  = 8;
constexpr int HD   = 128;
constexpr int OQKV = (NQ + 2 * NKV) * HD;   // 6144
constexpr int MROWS = Bk * Tk;              // 4096 token rows

// ---------------- workspace layout (f16 elements) ----------------
constexpr size_t OFF_XH    = 0;
constexpr size_t OFF_WQKV  = OFF_XH    + (size_t)MROWS * Hk;
constexpr size_t OFF_WPROJ = OFF_WQKV  + (size_t)OQKV  * Hk;
constexpr size_t OFF_Q     = OFF_WPROJ + (size_t)Hk * (NQ * HD);
constexpr size_t OFF_K     = OFF_Q     + (size_t)Bk * NQ  * Tk * HD;
constexpr size_t OFF_VT    = OFF_K     + (size_t)Bk * NKV * Tk * HD;
constexpr size_t OFF_ATTN  = OFF_VT    + (size_t)Bk * NKV * Tk * HD;
// total ~192 MB of f16 staging

// ---------------- WMMA helpers ----------------
__device__ __forceinline__ v8f wmma_f16(v16h a, v16h b, v8f c) {
  return __builtin_amdgcn_wmma_f32_16x16x32_f16(false, a, false, b, (short)0, c,
                                                false, false);
}

// A-matrix 16x32 f16 fragment: lane<16 rows hold K 0..7 / 16..23,
// lane>=16 hold K 8..15 / 24..31.  Operand row-major (M x K).
__device__ __forceinline__ v16h load_a_frag(const _Float16* __restrict__ base,
                                            int ld, int k0, int lane) {
  const int m = lane & 15;
  const int h = (lane >> 4) & 1;
  const _Float16* p = base + (size_t)m * ld + k0 + h * 8;
  h8 lo = *(const h8*)(p);
  h8 hi = *(const h8*)(p + 16);
  return __builtin_shufflevector(lo, hi, 0, 1, 2, 3, 4, 5, 6, 7,
                                         8, 9, 10, 11, 12, 13, 14, 15);
}

// B-matrix 32x16 f16 fragment: lane<16 (col n) holds K 0..7 / 16..23,
// lane>=16 holds K 8..15 / 24..31.  Operand stored (N x K) K-major.
__device__ __forceinline__ v16h load_b_frag(const _Float16* __restrict__ base,
                                            int ld, int k0, int lane) {
  const int n = lane & 15;
  const int h = (lane >> 4) & 1;
  const _Float16* p = base + (size_t)n * ld + k0 + h * 8;
  h8 lo = *(const h8*)(p);
  h8 hi = *(const h8*)(p + 16);
  return __builtin_shufflevector(lo, hi, 0, 1, 2, 3, 4, 5, 6, 7,
                                         8, 9, 10, 11, 12, 13, 14, 15);
}

// ---------------- CDNA5 async global->LDS copy ----------------
__device__ __forceinline__ void async_b128(unsigned lds_off,
                                           const _Float16* g) {
  asm volatile("global_load_async_to_lds_b128 %0, %1, off"
               :: "v"(lds_off), "v"(g) : "memory");
}
__device__ __forceinline__ void wait_async0() {
  asm volatile("s_wait_asynccnt 0" ::: "memory");
}

// ---------------- stage 0: f32 -> f16 ----------------
__global__ void cvt_f32_f16(const float* __restrict__ in,
                            _Float16* __restrict__ out, int n) {
  int i = blockIdx.x * blockDim.x + threadIdx.x;
  int stride = gridDim.x * blockDim.x;
  for (; i < n; i += stride) out[i] = (_Float16)in[i];
}

// ---------------- stage 1: QKV GEMM + RoPE + scatter ----------------
__global__ __launch_bounds__(256)
void gemm_qkv_rope(const _Float16* __restrict__ xh,
                   const _Float16* __restrict__ wh,
                   const float* __restrict__ fcos,
                   const float* __restrict__ fsin,
                   _Float16* __restrict__ qout,
                   _Float16* __restrict__ kout,
                   _Float16* __restrict__ vtout) {
  __shared__ _Float16 sT[8][16][16];           // per-wave V-tile transpose strip
  const int lane = threadIdx.x & 31;
  const int wave = threadIdx.x >> 5;
  const int wm = wave >> 2;
  const int wn = wave & 3;
  const int m0 = blockIdx.y * 64 + wm * 32;
  const int n0 = blockIdx.x * 256 + wn * 64;

  v8f acc[2][4];
#pragma unroll
  for (int i = 0; i < 2; ++i)
#pragma unroll
    for (int j = 0; j < 4; ++j) acc[i][j] = {};

  const int ln = lane & 15;
  const int g  = lane >> 4;

  for (int kk = 0; kk < Hk; kk += 32) {
    // unconditional prefetch of the streaming weight operand (4 K-steps ahead;
    // overrun stays inside the workspace, speculative prefetch is droppable)
#pragma unroll
    for (int j = 0; j < 4; ++j)
      __builtin_prefetch(wh + (size_t)(n0 + j * 16 + ln) * Hk + kk + 128, 0, 3);
    v16h a0 = load_a_frag(xh + (size_t)m0 * Hk, Hk, kk, lane);
    v16h a1 = load_a_frag(xh + (size_t)(m0 + 16) * Hk, Hk, kk, lane);
#pragma unroll
    for (int j = 0; j < 4; ++j) {
      v16h bj = load_b_frag(wh + (size_t)(n0 + j * 16) * Hk, Hk, kk, lane);
      acc[0][j] = wmma_f16(a0, bj, acc[0][j]);
      acc[1][j] = wmma_f16(a1, bj, acc[1][j]);
    }
  }

#pragma unroll
  for (int i = 0; i < 2; ++i) {
#pragma unroll
    for (int j = 0; j < 4; ++j) {
      const int cb   = n0 + j * 16;
      const int sub  = cb >> 7;
      const int kvh  = sub / 6;
      const int slot = sub % 6;
      const int qh_  = kvh * 4 + slot;
      if (slot == 5) {
        // V tile: transpose via LDS so stores run along t (coalesced 32B runs)
#pragma unroll
        for (int vr = 0; vr < 8; ++vr)
          sT[wave][vr + 8 * g][ln] = (_Float16)acc[i][j][vr];
        asm volatile("" ::: "memory");   // same-wave DS ops are in-order
        const int bb  = (m0 + i * 16) >> 10;
        const int tt0 = (m0 + i * 16) & 1023;
        const int d0  = cb & 127;
        _Float16* vrow =
            vtout + (((size_t)(bb * NKV + kvh)) * HD + d0) * Tk + tt0;
#pragma unroll
        for (int r = 0; r < 8; ++r) {
          const int dl = r + 8 * g;
          vrow[(size_t)dl * Tk + ln] = sT[wave][ln][dl];
        }
        asm volatile("" ::: "memory");
      } else {
        const int d = (cb & 127) + ln;
#pragma unroll
        for (int vr = 0; vr < 8; ++vr) {
          const int row = m0 + i * 16 + vr + 8 * g;
          const int bb  = row >> 10;
          const int tt  = row & 1023;
          float val = acc[i][j][vr];
          // RoPE via adjacent-lane exchange (pairs are even/odd d)
          float other = __shfl_xor(val, 1, 32);
          const int di = d >> 1;
          const float c = fcos[tt * (HD / 2) + di];
          const float s = fsin[tt * (HD / 2) + di];
          float r = ((d & 1) == 0) ? (val * c - other * s)
                                   : (other * s + val * c);
          if (slot == 4)
            kout[(((size_t)(bb * NKV + kvh)) * Tk + tt) * HD + d] = (_Float16)r;
          else
            qout[(((size_t)(bb * NQ + qh_)) * Tk + tt) * HD + d] = (_Float16)r;
        }
      }
    }
  }
}

// ---------------- stage 2: causal flash attention (GQA 4:1) ----------------
// Block = 4 waves = 64 query rows of one q-head.  K/V 32-key blocks staged
// cooperatively into LDS with CDNA5 async copies, double-buffered.
__global__ __launch_bounds__(128)
void attn_fwd(const _Float16* __restrict__ qin,
              const _Float16* __restrict__ kin,
              const _Float16* __restrict__ vtin,
              _Float16* __restrict__ aout) {
  __shared__ _Float16 skv[2][32 * HD + HD * 32];   // 2 x 16 KB K/V buffers
  __shared__ _Float16 sp[4][16 * 32];              // per-wave P strip

  const int tid  = threadIdx.x;
  const int lane = tid & 31;
  const int wave = tid >> 5;
  const int bh  = blockIdx.x;
  const int b   = bh / NQ;
  const int qh  = bh % NQ;
  const int kvh = qh >> 2;
  const int q0  = blockIdx.y * 64 + wave * 16;

  const _Float16* qbase = qin  + ((size_t)(b * NQ  + qh))  * Tk * HD;
  const _Float16* kbase = kin  + ((size_t)(b * NKV + kvh)) * Tk * HD;
  const _Float16* vtb   = vtin + ((size_t)(b * NKV + kvh)) * HD * Tk;
  _Float16* sw = sp[wave];

  auto stage = [&](int buf, int kb) {
    _Float16* kl = skv[buf];
    _Float16* vl = skv[buf] + 32 * HD;
    const _Float16* kg = kbase + (size_t)(kb * 32) * HD;
#pragma unroll
    for (int i = 0; i < 8; ++i) {            // K: 1024 16B chunks, 8/thread
      const int c = tid + i * 128;
      async_b128((unsigned)(uintptr_t)(kl + c * 8), kg + (size_t)c * 8);
    }
#pragma unroll
    for (int i = 0; i < 4; ++i) {            // V: 128 d-rows x 64B, 4/thread
      const int c   = tid + i * 128;
      const int row = c >> 2;
      const int sub = c & 3;
      async_b128((unsigned)(uintptr_t)(vl + row * 32 + sub * 8),
                 vtb + (size_t)row * Tk + kb * 32 + sub * 8);
    }
  };

  v16h qa[4];
#pragma unroll
  for (int c = 0; c < 4; ++c)
    qa[c] = load_a_frag(qbase + (size_t)q0 * HD, HD, c * 32, lane);

  const int ln = lane & 15;
  const int g  = lane >> 4;
  float mrow[8], lrow[8];
  v8f oacc[8];
#pragma unroll
  for (int vr = 0; vr < 8; ++vr) { mrow[vr] = -3.0e38f; lrow[vr] = 0.0f; }
#pragma unroll
  for (int j = 0; j < 8; ++j) oacc[j] = {};

  const float scale = 0.08838834764831845f;        // 1/sqrt(128)
  const int kbB = (blockIdx.y * 64 + 63) >> 5;     // block-wide max key block

  stage(0, 0);
  wait_async0();
  __syncthreads();

  for (int kb = 0; kb <= kbB; ++kb) {
    const int cur = kb & 1;
    if (kb < kbB) stage(cur ^ 1, kb + 1);          // overlap DMA with compute

    const _Float16* kbuf = skv[cur];
    const _Float16* vbuf = skv[cur] + 32 * HD;

    v8f c0 = {}, c1 = {};
#pragma unroll
    for (int kc = 0; kc < 4; ++kc) {
      v16h b0 = load_b_frag(kbuf,           HD, kc * 32, lane);
      v16h b1 = load_b_frag(kbuf + 16 * HD, HD, kc * 32, lane);
      c0 = wmma_f16(qa[kc], b0, c0);
      c1 = wmma_f16(qa[kc], b1, c1);
    }

    const int col0 = kb * 32 + ln;
#pragma unroll
    for (int vr = 0; vr < 8; ++vr) {
      const int row = q0 + vr + 8 * g;
      float s0 = c0[vr] * scale; if (col0 > row)      s0 = -3.0e38f;
      float s1 = c1[vr] * scale; if (col0 + 16 > row) s1 = -3.0e38f;
      float mx = fmaxf(s0, s1);
#pragma unroll
      for (int off = 1; off < 16; off <<= 1) mx = fmaxf(mx, __shfl_xor(mx, off, 32));
      const float mnew  = fmaxf(mrow[vr], mx);
      const float alpha = __expf(mrow[vr] - mnew);
      const float p0 = __expf(s0 - mnew);
      const float p1 = __expf(s1 - mnew);
      float ps = p0 + p1;
#pragma unroll
      for (int off = 1; off < 16; off <<= 1) ps += __shfl_xor(ps, off, 32);
      lrow[vr] = lrow[vr] * alpha + ps;
      mrow[vr] = mnew;
#pragma unroll
      for (int j = 0; j < 8; ++j) oacc[j][vr] *= alpha;
      sw[(vr + 8 * g) * 32 + ln]      = (_Float16)p0;
      sw[(vr + 8 * g) * 32 + ln + 16] = (_Float16)p1;
    }

    v16h pa = load_a_frag(sw, 32, 0, lane);
#pragma unroll
    for (int j = 0; j < 8; ++j) {
      v16h vb = load_b_frag(vbuf + (size_t)(j * 16) * 32, 32, 0, lane);
      oacc[j] = wmma_f16(pa, vb, oacc[j]);
    }

    if (kb < kbB) wait_async0();
    __syncthreads();
  }

#pragma unroll
  for (int vr = 0; vr < 8; ++vr) {
    const int row = q0 + vr + 8 * g;
    const float inv = 1.0f / lrow[vr];
    _Float16* orow = aout + ((size_t)(b * Tk + row)) * (NQ * HD) + qh * HD;
#pragma unroll
    for (int j = 0; j < 8; ++j)
      orow[j * 16 + ln] = (_Float16)(oacc[j][vr] * inv);
  }
}

// ---------------- stage 3: output projection ----------------
__global__ __launch_bounds__(256)
void gemm_proj(const _Float16* __restrict__ ah,
               const _Float16* __restrict__ wh,
               float* __restrict__ out) {
  const int lane = threadIdx.x & 31;
  const int wave = threadIdx.x >> 5;
  const int wm = wave >> 2;
  const int wn = wave & 3;
  const int m0 = blockIdx.y * 64 + wm * 32;
  const int n0 = blockIdx.x * 256 + wn * 64;
  constexpr int K = NQ * HD;

  v8f acc[2][4];
#pragma unroll
  for (int i = 0; i < 2; ++i)
#pragma unroll
    for (int j = 0; j < 4; ++j) acc[i][j] = {};

  const int ln = lane & 15;
  const int g  = lane >> 4;

  for (int kk = 0; kk < K; kk += 32) {
#pragma unroll
    for (int j = 0; j < 4; ++j)
      __builtin_prefetch(wh + (size_t)(n0 + j * 16 + ln) * K + kk + 128, 0, 3);
    v16h a0 = load_a_frag(ah + (size_t)m0 * K, K, kk, lane);
    v16h a1 = load_a_frag(ah + (size_t)(m0 + 16) * K, K, kk, lane);
#pragma unroll
    for (int j = 0; j < 4; ++j) {
      v16h bj = load_b_frag(wh + (size_t)(n0 + j * 16) * K, K, kk, lane);
      acc[0][j] = wmma_f16(a0, bj, acc[0][j]);
      acc[1][j] = wmma_f16(a1, bj, acc[1][j]);
    }
  }

#pragma unroll
  for (int i = 0; i < 2; ++i)
#pragma unroll
    for (int j = 0; j < 4; ++j) {
      const int cb = n0 + j * 16;
#pragma unroll
      for (int vr = 0; vr < 8; ++vr) {
        const int row = m0 + i * 16 + vr + 8 * g;
        out[(size_t)row * Hk + cb + ln] = acc[i][j][vr];
      }
    }
}

// ---------------- host-side launcher ----------------
extern "C" void kernel_launch(void* const* d_in, const int* in_sizes, int n_in,
                              void* d_out, int out_size, void* d_ws, size_t ws_size,
                              hipStream_t stream) {
  const float* x      = (const float*)d_in[0];
  const float* w_qkv  = (const float*)d_in[1];
  const float* w_proj = (const float*)d_in[2];
  const float* fcos   = (const float*)d_in[3];
  const float* fsin   = (const float*)d_in[4];

  _Float16* ws     = (_Float16*)d_ws;
  _Float16* xh     = ws + OFF_XH;
  _Float16* wqkvh  = ws + OFF_WQKV;
  _Float16* wprojh = ws + OFF_WPROJ;
  _Float16* qh     = ws + OFF_Q;
  _Float16* kh     = ws + OFF_K;
  _Float16* vth    = ws + OFF_VT;
  _Float16* attnh  = ws + OFF_ATTN;

  cvt_f32_f16<<<2048, 256, 0, stream>>>(x,      xh,     MROWS * Hk);
  cvt_f32_f16<<<2048, 256, 0, stream>>>(w_qkv,  wqkvh,  OQKV * Hk);
  cvt_f32_f16<<<2048, 256, 0, stream>>>(w_proj, wprojh, Hk * NQ * HD);

  dim3 g1(OQKV / 256, MROWS / 64);
  gemm_qkv_rope<<<g1, 256, 0, stream>>>(xh, wqkvh, fcos, fsin, qh, kh, vth);

  dim3 g2(Bk * NQ, Tk / 64);
  attn_fwd<<<g2, 128, 0, stream>>>(qh, kh, vth, attnh);

  dim3 g3(Hk / 256, MROWS / 64);
  gemm_proj<<<g3, 256, 0, stream>>>(attnh, wprojh, (float*)d_out);
}
